// LocalizedSelfAttention_36730560315526
// MI455X (gfx1250) — matrix operand order repaired and checked by reading, not compiled, not run on hardware
//
#include <hip/hip_runtime.h>
#include <hip/hip_bf16.h>

// ---------- types ----------
typedef __bf16 bf16;
typedef __attribute__((ext_vector_type(16))) __bf16 v16bf;
typedef __attribute__((ext_vector_type(8)))  __bf16 v8bf;
typedef __attribute__((ext_vector_type(4)))  __bf16 v4bf;
typedef __attribute__((ext_vector_type(8)))  float  v8f;

union Frag { v16bf v; v8bf h[2]; };

__device__ __forceinline__ v8f wmma_bf16(const v16bf& a, const v16bf& b, const v8f& c) {
  return __builtin_amdgcn_wmma_f32_16x16x32_bf16(false, a, false, b, (short)0, c, false, false);
}

__device__ __forceinline__ float rowmax16(float v) {
  v = fmaxf(v, __shfl_xor(v, 1, 32));
  v = fmaxf(v, __shfl_xor(v, 2, 32));
  v = fmaxf(v, __shfl_xor(v, 4, 32));
  v = fmaxf(v, __shfl_xor(v, 8, 32));
  return v;
}
__device__ __forceinline__ float rowsum16(float v) {
  v += __shfl_xor(v, 1, 32);
  v += __shfl_xor(v, 2, 32);
  v += __shfl_xor(v, 4, 32);
  v += __shfl_xor(v, 8, 32);
  return v;
}
__device__ __forceinline__ float wave_sum(float v) {
  #pragma unroll
  for (int m = 1; m < 32; m <<= 1) v += __shfl_xor(v, m, 32);
  return v;
}

// ---------- problem constants ----------
#define BB 4
#define SS 4096
#define DD 1024
#define HH 16
#define KK 64
#define NW 8
#define SW 512            // window length
#define MTOT (BB*SS)      // 16384 rows
#define NQK 2048          // fused Q|K projection width

// ---------- conversion / packing (one-time prep) ----------
__global__ __launch_bounds__(256) void cvt_f32_bf16_kernel(const float* __restrict__ in,
                                                           bf16* __restrict__ out) {
  int i = blockIdx.x * 256 + threadIdx.x;          // one float4 per thread
  float4 f = ((const float4*)in)[i];
  v4bf o; o[0] = (bf16)f.x; o[1] = (bf16)f.y; o[2] = (bf16)f.z; o[3] = (bf16)f.w;
  ((v4bf*)out)[i] = o;
}

// Wt[c][d] = {Wq|Wk}[d][c]  (pre-transposed fused QK weight) + packed bias
__global__ __launch_bounds__(256) void pack_wqk_t_kernel(const float* __restrict__ Wq,
                                                         const float* __restrict__ Wk,
                                                         const float* __restrict__ bq,
                                                         const float* __restrict__ bk,
                                                         bf16* __restrict__ Wt,
                                                         float* __restrict__ bias) {
  int i = blockIdx.x * 256 + threadIdx.x;          // over 2048*1024
  int c = i >> 10, d = i & 1023;
  const float* src = (c < 1024) ? Wq : Wk;
  Wt[i] = (bf16)src[d * 1024 + (c & 1023)];
  if (i < NQK) bias[i] = (i < 1024) ? bq[i] : bk[i - 1024];
}

// Wt[c][d] = W[d][c] for a 1024x1024 weight; optional bias copy
__global__ __launch_bounds__(256) void pack_w_t_kernel(const float* __restrict__ W,
                                                       const float* __restrict__ b,
                                                       bf16* __restrict__ Wt,
                                                       float* __restrict__ bias) {
  int i = blockIdx.x * 256 + threadIdx.x;          // over 1024*1024
  int c = i >> 10, d = i & 1023;
  Wt[i] = (bf16)W[d * 1024 + c];
  if (i < 1024 && bias) bias[i] = b[i];
}

// ---------- generic bf16 WMMA GEMM: C[M,N] = A[M,Kd] * Wt[N,Kd]^T + bias ----------
// block tile 128x128, 8 waves (4M x 2N), K-step 64, double-buffered LDS.
// EP: 0 = f32 row-major, 1 = bf16 row-major, 2 = bf16 V-transposed [g,h,dim,tok]
template<int EP>
__global__ __launch_bounds__(256)
void gemm_bf16_kernel(const bf16* __restrict__ A, const bf16* __restrict__ Wt,
                      const float* __restrict__ bias, void* __restrict__ Cout,
                      int M, int N, int Kd) {
  __shared__ __align__(16) bf16 As[2][128][72];   // rows x (64 K + pad)
  __shared__ __align__(16) bf16 Bt[2][128][72];   // cols x (64 K + pad)
  const int tid  = threadIdx.x;
  const int lane = tid & 31;
  const int wv   = tid >> 5;
  const int wm   = wv & 3;          // 0..3 : 32-row strip
  const int wn   = wv >> 2;         // 0..1 : 64-col strip
  const int half = lane >> 4;       // 0/1
  const int l16  = lane & 15;
  const int rowBase = blockIdx.y * 128;
  const int colBase = blockIdx.x * 128;

  v8f acc[2][4] = {};

  auto stage = [&](int buf, int k0) {
    #pragma unroll
    for (int it = 0; it < 4; ++it) {
      int idx = tid + it * 256;
      int r = idx >> 3;              // 0..127
      int c = (idx & 7) * 8;         // 0..56
      *(v8bf*)&As[buf][r][c] = *(const v8bf*)(A  + (size_t)(rowBase + r) * Kd + k0 + c);
      *(v8bf*)&Bt[buf][r][c] = *(const v8bf*)(Wt + (size_t)(colBase + r) * Kd + k0 + c);
    }
  };

  stage(0, 0);
  __syncthreads();
  const int iters = Kd >> 6;
  for (int it = 0; it < iters; ++it) {
    int cur = it & 1;
    if (it + 1 < iters) stage(cur ^ 1, (it + 1) * 64);   // overlap next tile with WMMAs
    #pragma unroll
    for (int kk = 0; kk < 64; kk += 32) {
      Frag a[2], b[4];
      #pragma unroll
      for (int mf = 0; mf < 2; ++mf) {
        int r  = wm * 32 + mf * 16 + l16;
        int cb = kk + half * 8;
        a[mf].h[0] = *(const v8bf*)&As[cur][r][cb];
        a[mf].h[1] = *(const v8bf*)&As[cur][r][cb + 16];
      }
      #pragma unroll
      for (int nf = 0; nf < 4; ++nf) {
        int n  = wn * 64 + nf * 16 + l16;
        int kb = kk + half * 8;
        b[nf].h[0] = *(const v8bf*)&Bt[cur][n][kb];
        b[nf].h[1] = *(const v8bf*)&Bt[cur][n][kb + 16];
      }
      #pragma unroll
      for (int mf = 0; mf < 2; ++mf)
        #pragma unroll
        for (int nf = 0; nf < 4; ++nf)
          acc[mf][nf] = wmma_bf16(a[mf].v, b[nf].v, acc[mf][nf]);
    }
    __syncthreads();
  }

  // epilogue: + bias, store
  #pragma unroll
  for (int nf = 0; nf < 4; ++nf) {
    int col = colBase + wn * 64 + nf * 16 + l16;
    float bvv = bias ? bias[col] : 0.0f;
    #pragma unroll
    for (int mf = 0; mf < 2; ++mf) {
      #pragma unroll
      for (int j = 0; j < 8; ++j) {
        int row = rowBase + wm * 32 + mf * 16 + j + half * 8;
        float v = acc[mf][nf][j] + bvv;
        if (EP == 0) {
          ((float*)Cout)[(size_t)row * N + col] = v;
        } else if (EP == 1) {
          ((bf16*)Cout)[(size_t)row * N + col] = (bf16)v;
        } else {
          // V output, transposed: [(g*16+h)*64+dim][512 tokens]
          int g = row >> 9, tw = row & 511;
          int h = col >> 6, d = col & 63;
          ((bf16*)Cout)[(((size_t)(g * HH + h) * KK + d) << 9) + tw] = (bf16)v;
        }
      }
    }
  }
}

// ---------- windowed flash attention ----------
// grid: (qtile 0..3, window*head 0..511); block 256 (8 waves, 16 q-rows each)
// Q,K from qk[tok][2048] (Q cols 0..1023, K cols 1024..2047); V from vT[g,h,dim,tok]
// Double-buffered K/V staging: one barrier per 64-key step.
__global__ __launch_bounds__(256)
void attn_kernel(const bf16* __restrict__ qk, const bf16* __restrict__ vT,
                 bf16* __restrict__ ctx) {
  __shared__ __align__(16) bf16 Kt[2][64][72];     // keys x (64 dims + pad)
  __shared__ __align__(16) bf16 Vt[2][64][72];     // dims x (64 keys + pad)
  __shared__ __align__(16) bf16 Pt[8][16][64];     // per-wave P scratch (16 q x 64 keys)
  const int tid  = threadIdx.x;
  const int lane = tid & 31;
  const int wv   = tid >> 5;
  const int half = lane >> 4;
  const int l16  = lane & 15;
  const int g = blockIdx.y >> 4;                   // window 0..31 (b*8+w)
  const int h = blockIdx.y & 15;
  const int qtile = blockIdx.x;                    // 0..3
  const size_t tokBase = (size_t)g * SW;
  const bf16* vbase = vT + ((size_t)(g * HH + h) * KK << 9);   // [dim][512]

  auto stage = [&](int buf, int s0) {
    #pragma unroll
    for (int it = 0; it < 2; ++it) {
      int idx = tid + it * 256;
      int r = idx >> 3;               // 0..63
      int c = (idx & 7) * 8;          // 0..56
      *(v8bf*)&Kt[buf][r][c] = *(const v8bf*)(qk + (tokBase + s0 + r) * NQK + 1024 + h * KK + c);
      *(v8bf*)&Vt[buf][r][c] = *(const v8bf*)(vbase + ((size_t)r << 9) + s0 + c);
    }
  };

  // Q fragments (d = 64 -> two 16x32 A fragments), straight from global
  Frag qf[2];
  {
    const int qrow = qtile * 128 + wv * 16 + l16;
    const bf16* qp = qk + (tokBase + qrow) * NQK + h * KK;
    #pragma unroll
    for (int c = 0; c < 2; ++c) {
      int base = c * 32 + half * 8;
      qf[c].h[0] = *(const v8bf*)(qp + base);
      qf[c].h[1] = *(const v8bf*)(qp + base + 16);
    }
  }

  float m[8], l[8];
  v8f acc[4] = {};
  #pragma unroll
  for (int j = 0; j < 8; ++j) { m[j] = -1.0e30f; l[j] = 0.0f; }

  stage(0, 0);
  __syncthreads();
  const int iters = SW / 64;                       // 8
  for (int it = 0; it < iters; ++it) {
    int cur = it & 1;
    if (it + 1 < iters) stage(cur ^ 1, (it + 1) * 64);   // overlap next K/V tile

    // scores: four 16-key subtiles, each = two chained WMMAs over d=64
    v8f sc[4];
    #pragma unroll
    for (int sub = 0; sub < 4; ++sub) {
      Frag kb0, kb1;
      int key = sub * 16 + l16;
      int kb  = half * 8;
      kb0.h[0] = *(const v8bf*)&Kt[cur][key][kb];
      kb0.h[1] = *(const v8bf*)&Kt[cur][key][kb + 16];
      kb1.h[0] = *(const v8bf*)&Kt[cur][key][32 + kb];
      kb1.h[1] = *(const v8bf*)&Kt[cur][key][32 + kb + 16];
      v8f z = {};
      v8f t = wmma_bf16(qf[0].v, kb0.v, z);
      sc[sub] = wmma_bf16(qf[1].v, kb1.v, t);
    }

    // one online-softmax update per 64 keys (row shared by a 16-lane half)
    #pragma unroll
    for (int j = 0; j < 8; ++j) {
      float s0v = sc[0][j] * 0.125f;          // 1/sqrt(64)
      float s1v = sc[1][j] * 0.125f;
      float s2v = sc[2][j] * 0.125f;
      float s3v = sc[3][j] * 0.125f;
      float tm = rowmax16(fmaxf(fmaxf(s0v, s1v), fmaxf(s2v, s3v)));
      float mn = fmaxf(m[j], tm);
      float corr = __expf(m[j] - mn);
      float p0 = __expf(s0v - mn);
      float p1 = __expf(s1v - mn);
      float p2 = __expf(s2v - mn);
      float p3 = __expf(s3v - mn);
      float rs = rowsum16((p0 + p1) + (p2 + p3));
      l[j] = l[j] * corr + rs;
      m[j] = mn;
      #pragma unroll
      for (int nf = 0; nf < 4; ++nf) acc[nf][j] *= corr;
      int prow = j + half * 8;
      Pt[wv][prow][l16]      = (bf16)p0;
      Pt[wv][prow][16 + l16] = (bf16)p1;
      Pt[wv][prow][32 + l16] = (bf16)p2;
      Pt[wv][prow][48 + l16] = (bf16)p3;
    }
    // intra-wave LDS turnaround (per-wave buffer; LDS is in-order per wave)
    __builtin_amdgcn_wave_barrier();
    asm volatile("s_wait_dscnt 0" ::: "memory");

    // P (16x64) as two A fragments; PV = 8 WMMAs into the 64-wide context
    Frag pf[2];
    #pragma unroll
    for (int c = 0; c < 2; ++c) {
      pf[c].h[0] = *(const v8bf*)&Pt[wv][l16][c * 32 + half * 8];
      pf[c].h[1] = *(const v8bf*)&Pt[wv][l16][c * 32 + 16 + half * 8];
    }
    #pragma unroll
    for (int nf = 0; nf < 4; ++nf) {
      int dim = nf * 16 + l16;
      #pragma unroll
      for (int c = 0; c < 2; ++c) {
        Frag vf;
        int kb = c * 32 + half * 8;
        vf.h[0] = *(const v8bf*)&Vt[cur][dim][kb];
        vf.h[1] = *(const v8bf*)&Vt[cur][dim][kb + 16];
        acc[nf] = wmma_bf16(pf[c].v, vf.v, acc[nf]);
      }
    }
    __syncthreads();   // all waves done reading 'cur'; next iter stages into it
  }

  // finalize: divide by softmax denominator, store bf16 context
  #pragma unroll
  for (int j = 0; j < 8; ++j) {
    float inv = 1.0f / l[j];
    int row = qtile * 128 + wv * 16 + j + half * 8;
    size_t tok = tokBase + row;
    #pragma unroll
    for (int nf = 0; nf < 4; ++nf) {
      float v = acc[nf][j] * inv;
      ctx[tok * (size_t)DD + h * KK + nf * 16 + l16] = (bf16)v;
    }
  }
}

// ---------- residual + LayerNorm (eps = 1e-3) ----------
__global__ __launch_bounds__(256)
void ln_kernel(const float* __restrict__ x, const float* __restrict__ attn,
               const float* __restrict__ gamma, const float* __restrict__ beta,
               float* __restrict__ out) {
  __shared__ float rs[8], rq[8];
  const int tid = threadIdx.x;
  const int lane = tid & 31, wv = tid >> 5;
  const size_t base = (size_t)blockIdx.x * DD;
  float4 xv = ((const float4*)(x + base))[tid];
  float4 av = ((const float4*)(attn + base))[tid];
  float y0 = xv.x + av.x, y1 = xv.y + av.y, y2 = xv.z + av.z, y3 = xv.w + av.w;
  float s = y0 + y1 + y2 + y3;
  float q = y0 * y0 + y1 * y1 + y2 * y2 + y3 * y3;
  s = wave_sum(s); q = wave_sum(q);
  if (lane == 0) { rs[wv] = s; rq[wv] = q; }
  __syncthreads();
  float ts = 0.f, tq = 0.f;
  #pragma unroll
  for (int i = 0; i < 8; ++i) { ts += rs[i]; tq += rq[i]; }
  float mu  = ts * (1.0f / DD);
  float var = tq * (1.0f / DD) - mu * mu;
  float r = rsqrtf(var + 1e-3f);
  float4 gv = ((const float4*)gamma)[tid];
  float4 bv = ((const float4*)beta)[tid];
  float4 o;
  o.x = (y0 - mu) * r * gv.x + bv.x;
  o.y = (y1 - mu) * r * gv.y + bv.y;
  o.z = (y2 - mu) * r * gv.z + bv.z;
  o.w = (y3 - mu) * r * gv.w + bv.w;
  ((float4*)(out + base))[tid] = o;
}

// ---------- host launcher ----------
extern "C" void kernel_launch(void* const* d_in, const int* in_sizes, int n_in,
                              void* d_out, int out_size, void* d_ws, size_t ws_size,
                              hipStream_t stream) {
  const float* x     = (const float*)d_in[0];
  const float* Wq    = (const float*)d_in[1];
  const float* bq    = (const float*)d_in[2];
  const float* Wk    = (const float*)d_in[3];
  const float* bk    = (const float*)d_in[4];
  const float* Wv    = (const float*)d_in[5];
  const float* bv    = (const float*)d_in[6];
  const float* Wo    = (const float*)d_in[7];
  const float* bo    = (const float*)d_in[8];
  const float* gamma = (const float*)d_in[9];
  const float* beta  = (const float*)d_in[10];
  float* out = (float*)d_out;

  // workspace carve-up (256B aligned)
  char* ws = (char*)d_ws;
  size_t off = 0;
  auto alloc = [&](size_t bytes) { void* p = ws + off; off = (off + bytes + 255) & ~(size_t)255; return p; };
  bf16*  x_bf     = (bf16*)alloc((size_t)MTOT * DD * 2);          // 32 MB
  bf16*  Wqk_t    = (bf16*)alloc((size_t)NQK * DD * 2);           // 4 MB
  bf16*  Wv_t     = (bf16*)alloc((size_t)DD * DD * 2);            // 2 MB
  bf16*  Wo_t     = (bf16*)alloc((size_t)DD * DD * 2);            // 2 MB
  float* bias_qk  = (float*)alloc((size_t)NQK * 4);
  float* bias_v   = (float*)alloc((size_t)DD * 4);
  bf16*  qk_bf    = (bf16*)alloc((size_t)MTOT * NQK * 2);         // 64 MB
  bf16*  vT       = (bf16*)alloc((size_t)MTOT * DD * 2);          // 32 MB
  bf16*  ctx_bf   = (bf16*)alloc((size_t)MTOT * DD * 2);          // 32 MB
  float* attn_f32 = (float*)alloc((size_t)MTOT * DD * 4);         // 64 MB

  // 1) conversions / packing (pre-transposed weights)
  cvt_f32_bf16_kernel<<<(MTOT * DD) / 1024, 256, 0, stream>>>(x, x_bf);
  pack_wqk_t_kernel<<<(NQK * DD) / 256, 256, 0, stream>>>(Wq, Wk, bq, bk, Wqk_t, bias_qk);
  pack_w_t_kernel<<<(DD * DD) / 256, 256, 0, stream>>>(Wv, bv, Wv_t, bias_v);
  pack_w_t_kernel<<<(DD * DD) / 256, 256, 0, stream>>>(Wo, nullptr, Wo_t, nullptr);
  // 2) fused Q|K projection: [16384 x 1024] x [1024 x 2048] -> bf16 row-major
  gemm_bf16_kernel<1><<<dim3(NQK / 128, MTOT / 128), 256, 0, stream>>>(
      x_bf, Wqk_t, bias_qk, (void*)qk_bf, MTOT, NQK, DD);
  // 3) V projection: [16384 x 1024] x [1024 x 1024] -> bf16, transposed [g,h,dim,tok]
  gemm_bf16_kernel<2><<<dim3(DD / 128, MTOT / 128), 256, 0, stream>>>(
      x_bf, Wv_t, bias_v, (void*)vT, MTOT, DD, DD);
  // 4) windowed attention -> bf16 context [16384 x 1024]
  attn_kernel<<<dim3(SW / 128, BB * NW * HH), 256, 0, stream>>>(qk_bf, vT, ctx_bf);
  // 5) output projection: [16384 x 1024] x [1024 x 1024] + bo -> f32
  gemm_bf16_kernel<0><<<dim3(DD / 128, MTOT / 128), 256, 0, stream>>>(
      ctx_bf, Wo_t, bo, (void*)attn_f32, MTOT, DD, DD);
  // 6) residual + layernorm
  ln_kernel<<<MTOT, 256, 0, stream>>>(x, attn_f32, gamma, beta, out);
}
